// INNModel_16947940950461
// MI455X (gfx1250) — matrix operand order repaired and checked by reading, not compiled, not run on hardware
//
#include <hip/hip_runtime.h>
#include <hip/hip_bf16.h>

typedef __attribute__((ext_vector_type(16))) _Float16 v16h;
typedef __attribute__((ext_vector_type(8)))  float    v8f;
typedef __attribute__((ext_vector_type(4)))  unsigned int u32x4;
typedef __attribute__((ext_vector_type(8)))  int    i32x8;
typedef __attribute__((ext_vector_type(4)))  int    i32x4;

#define SEQ   4096
#define NENT  512
#define NREL  1024
#define EMB   512
#define HID   512
#define H2    1024
#define REL   256
#define NTOT  (NENT + NREL)   // 1536
#define G4    (4 * HID)       // 2048
#define WE_N  (4 * H2)        // 4096
#define VOCAB 32000
#define NELEM 64

// ---------------------------------------------------------------------------
// Gather rows of an embedding table (with index clamp) into a dense f32 matrix
// ---------------------------------------------------------------------------
__global__ void gather_rows(const int* __restrict__ idxs,
                            const float* __restrict__ table,
                            float* __restrict__ out,
                            int rows, int width, int table_rows) {
  int idx = blockIdx.x * blockDim.x + threadIdx.x;
  if (idx >= rows * width) return;
  int r = idx / width, c = idx - r * width;
  int t = idxs[r];
  t = t < 0 ? 0 : (t > table_rows - 1 ? table_rows - 1 : t);
  out[idx] = table[(size_t)t * width + c];
}

// ---------------------------------------------------------------------------
// Pack a row-major f32 [M,K] matrix into WMMA A-fragment order (16-bit A,
// 16x32 tiles, per ISA 7.12.2): dst[((mt*KT+kt)*32 + lane)*16 + j]
// ---------------------------------------------------------------------------
__global__ void pack_a16(const float* __restrict__ src, _Float16* __restrict__ dst,
                         int M, int K) {
  int idx = blockIdx.x * blockDim.x + threadIdx.x;
  if (idx >= M * K) return;
  int j    = idx & 15;
  int lane = (idx >> 4) & 31;
  int rest = idx >> 9;           // / (16*32)
  int KT   = K >> 5;
  int kt   = rest % KT;
  int mt   = rest / KT;
  int m    = mt * 16 + (lane & 15);
  int hi   = lane >> 4;
  int v    = j >> 1, t = j & 1;
  int kin  = (v < 4) ? (hi * 8 + v * 2 + t) : (16 + hi * 8 + (v - 4) * 2 + t);
  int k    = kt * 32 + kin;
  dst[idx] = (_Float16)src[(size_t)m * K + k];
}

// ---------------------------------------------------------------------------
// Pack B (K x N logical) into WMMA B-fragment order (16-bit B, 32x16 tiles):
// lane<16 -> K=0..7,16..23 of col n, lane>=16 -> K=8..15,24..31.
// transposed=1: src is row-major [N,K] (i.e. B = src^T, used for x @ W^T).
// ---------------------------------------------------------------------------
__global__ void pack_b16(const float* __restrict__ src, _Float16* __restrict__ dst,
                         int K, int Nn, int transposed) {
  int idx = blockIdx.x * blockDim.x + threadIdx.x;
  if (idx >= K * Nn) return;
  int j    = idx & 15;
  int lane = (idx >> 4) & 31;
  int rest = idx >> 9;
  int KT   = K >> 5;
  int kt   = rest % KT;
  int nt   = rest / KT;
  int n    = nt * 16 + (lane & 15);
  int k    = kt * 32 + ((j < 8) ? ((lane >> 4) * 8 + j) : (16 + (lane >> 4) * 8 + (j - 8)));
  float v  = transposed ? src[(size_t)n * K + k] : src[(size_t)k * Nn + n];
  dst[idx] = (_Float16)v;
}

// ---------------------------------------------------------------------------
// Register-blocked pre-packed WMMA GEMM: C = A*B (+bias). Each wave owns a
// 64x64 output block = 4x4 WMMA tiles: per K-step it loads 4 A-fragments and
// 4 B-fragments (contiguous 32B each) and issues 16 v_wmma_f32_16x16x32_f16,
// giving 4x fragment reuse and 16 independent accumulator chains. Grids are
// sized exactly so EXEC is all-ones at every WMMA (ISA 7.12 requirement).
// ---------------------------------------------------------------------------
__global__ void __launch_bounds__(256)
wmma_gemm4x4(const _Float16* __restrict__ Ap, const _Float16* __restrict__ Bp,
             float* __restrict__ C, const float* __restrict__ bias,
             int NT4, int KT, int Nn) {
  int wave = (int)((blockIdx.x * blockDim.x + threadIdx.x) >> 5);
  int lane = threadIdx.x & 31;
  int mt4 = wave / NT4, nt4 = wave % NT4;

  const _Float16* aB[4];
  const _Float16* bB[4];
#pragma unroll
  for (int i = 0; i < 4; ++i) {
    aB[i] = Ap + (((size_t)(mt4 * 4 + i) * KT) << 9) + (lane << 4);
    bB[i] = Bp + (((size_t)(nt4 * 4 + i) * KT) << 9) + (lane << 4);
  }

  v8f acc[4][4] = {};
  for (int kt = 0; kt < KT; ++kt) {
    v16h a[4], b[4];
#pragma unroll
    for (int i = 0; i < 4; ++i) a[i] = *(const v16h*)(aB[i] + ((size_t)kt << 9));
#pragma unroll
    for (int j = 0; j < 4; ++j) b[j] = *(const v16h*)(bB[j] + ((size_t)kt << 9));
    if (kt + 1 < KT) {  // global_prefetch_b8 hints for next K-fragments
      __builtin_prefetch(aB[0] + ((size_t)(kt + 1) << 9), 0, 3);
      __builtin_prefetch(bB[0] + ((size_t)(kt + 1) << 9), 0, 3);
    }
#pragma unroll
    for (int i = 0; i < 4; ++i)
#pragma unroll
      for (int j = 0; j < 4; ++j)
        acc[i][j] = __builtin_amdgcn_wmma_f32_16x16x32_f16(
            false, a[i], false, b[j], (short)0, acc[i][j], false, false);
  }

#pragma unroll
  for (int i = 0; i < 4; ++i) {
    int rbase = (mt4 * 4 + i) * 16 + ((lane >> 4) << 3);
#pragma unroll
    for (int j = 0; j < 4; ++j) {
      int col  = (nt4 * 4 + j) * 16 + (lane & 15);
      float bv = bias ? bias[col] : 0.0f;
#pragma unroll
      for (int v = 0; v < 8; ++v)
        C[(size_t)(rbase + v) * Nn + col] = acc[i][j][v] + bv;
    }
  }
}

// ---------------------------------------------------------------------------
// TDM tile copy: one wave per 8x1024-f32 tile. Builds a 2D Tensor DMA
// descriptor (D# group0/group1, ISA 08_async_tensor 8.3/8.4), then
// tensor_load_to_lds -> s_wait_tensorcnt 0 -> tensor_store_from_lds.
// ---------------------------------------------------------------------------
#define TDM_ROWS 8
#define TDM_COLS H2

#if defined(__has_builtin)
#if __has_builtin(__builtin_amdgcn_tensor_load_to_lds) && \
    __has_builtin(__builtin_amdgcn_tensor_store_from_lds) && \
    __has_builtin(__builtin_amdgcn_s_wait_tensorcnt)
#define HAVE_TDM 1
#endif
#endif

__device__ inline void tdm_make_desc(unsigned g0[4], unsigned g1[8],
                                     unsigned long long gaddr, unsigned ldsaddr,
                                     unsigned dim0, unsigned dim1,
                                     unsigned tile0, unsigned tile1,
                                     unsigned long long stride0) {
  // group0: count=1 | lds_addr | global_addr[56:0] | type=2
  g0[0] = 1u;
  g0[1] = ldsaddr;
  g0[2] = (unsigned)(gaddr & 0xFFFFFFFFu);
  g0[3] = (unsigned)((gaddr >> 32) & 0x1FFFFFFu) | (2u << 30);
  // group1: wg_mask=0, data_size=2 (4B), dims/tiles/stride
  g1[0] = (2u << 16);
  g1[1] = (dim0 & 0xFFFFu) << 16;                       // tensor_dim0[15:0]
  g1[2] = (dim0 >> 16) | ((dim1 & 0xFFFFu) << 16);      // dim0[31:16] | dim1[15:0]
  g1[3] = (dim1 >> 16) | ((tile0 & 0xFFFFu) << 16);     // dim1[31:16] | tile_dim0
  g1[4] = (tile1 & 0xFFFFu);                            // tile_dim1 (tile_dim2=0)
  g1[5] = (unsigned)(stride0 & 0xFFFFFFFFu);            // tensor_dim0_stride[31:0]
  g1[6] = (unsigned)((stride0 >> 32) & 0xFFFFu);        // stride0[47:32] (dim1_stride=0)
  g1[7] = 0u;
}

__global__ void __launch_bounds__(32)
tdm_copy_2d(const float* __restrict__ src, float* __restrict__ dst,
            int nrows, int ncols) {
  __shared__ float lbuf[TDM_ROWS * TDM_COLS];
  size_t row0 = (size_t)blockIdx.x * TDM_ROWS;
#if defined(HAVE_TDM)
  unsigned lds = (unsigned)(size_t)lbuf;
  unsigned long long gs = (unsigned long long)(size_t)(src + row0 * ncols);
  unsigned long long gd = (unsigned long long)(size_t)(dst + row0 * ncols);
  unsigned g0[4], g1[8];
  // load tile: global -> LDS
  tdm_make_desc(g0, g1, gs, lds, (unsigned)ncols, (unsigned)nrows,
                (unsigned)ncols, TDM_ROWS, (unsigned long long)ncols);
  u32x4 G0 = {g0[0], g0[1], g0[2], g0[3]};
  i32x8 G1 = {(int)g1[0], (int)g1[1], (int)g1[2], (int)g1[3],
              (int)g1[4], (int)g1[5], (int)g1[6], (int)g1[7]};
  i32x4 Z4 = {0, 0, 0, 0};
#if defined(__clang_major__) && (__clang_major__ >= 23)
  i32x8 Z8 = {0, 0, 0, 0, 0, 0, 0, 0};
  __builtin_amdgcn_tensor_load_to_lds(G0, G1, Z4, Z4, Z8, 0);
#else
  __builtin_amdgcn_tensor_load_to_lds(G0, G1, Z4, Z4, 0);
#endif
  __builtin_amdgcn_s_wait_tensorcnt(0);
  // store tile: LDS -> global
  tdm_make_desc(g0, g1, gd, lds, (unsigned)ncols, (unsigned)nrows,
                (unsigned)ncols, TDM_ROWS, (unsigned long long)ncols);
  u32x4 H0 = {g0[0], g0[1], g0[2], g0[3]};
  i32x8 H1 = {(int)g1[0], (int)g1[1], (int)g1[2], (int)g1[3],
              (int)g1[4], (int)g1[5], (int)g1[6], (int)g1[7]};
#if defined(__clang_major__) && (__clang_major__ >= 23)
  __builtin_amdgcn_tensor_store_from_lds(H0, H1, Z4, Z4, Z8, 0);
#else
  __builtin_amdgcn_tensor_store_from_lds(H0, H1, Z4, Z4, 0);
#endif
  __builtin_amdgcn_s_wait_tensorcnt(0);
#else
  // Fallback: plain wave copy
  for (int r = 0; r < TDM_ROWS; ++r)
    for (int c = threadIdx.x; c < ncols; c += 32)
      dst[(row0 + r) * ncols + c] = src[(row0 + r) * ncols + c];
  (void)lbuf;
#endif
}

// ---------------------------------------------------------------------------
// Sequential LSTM scan for one direction. Xg holds the precomputed x@W_ih^T.
// h, c live in LDS; 1024 threads each own 2 of the 2048 gate lanes.
// ---------------------------------------------------------------------------
__global__ void __launch_bounds__(1024)
lstm_scan(const float* __restrict__ Xg, const float* __restrict__ Whh,
          const float* __restrict__ bias, float* __restrict__ blstm, int reverse) {
  __shared__ float h[HID];
  __shared__ float c[HID];
  __shared__ float g[G4];
  int tid = threadIdx.x;
  if (tid < HID) { h[tid] = 0.f; c[tid] = 0.f; }
  __syncthreads();
  for (int step = 0; step < SEQ; ++step) {
    int t = reverse ? (SEQ - 1 - step) : step;
    const float* xr = Xg + (size_t)t * G4;
#pragma unroll
    for (int jj = 0; jj < 2; ++jj) {
      int j = tid + jj * 1024;
      float acc = xr[j] + bias[j];
      const float* w = Whh + (size_t)j * HID;
      for (int e = 0; e < HID; ++e) acc += w[e] * h[e];
      g[j] = acc;
    }
    __syncthreads();
    if (tid < HID) {
      float ig = 1.f / (1.f + __expf(-g[tid]));
      float fg = 1.f / (1.f + __expf(-g[HID + tid]));
      float gg = tanhf(g[2 * HID + tid]);
      float og = 1.f / (1.f + __expf(-g[3 * HID + tid]));
      float cn = fg * c[tid] + ig * gg;
      c[tid] = cn;
      float hn = og * tanhf(cn);
      h[tid] = hn;
      blstm[(size_t)t * H2 + (reverse ? HID : 0) + tid] = hn;
    }
    __syncthreads();
  }
}

// ---------------------------------------------------------------------------
// Attention scores: scores[s] = blstm[s] . attn_w + attn_b
// ---------------------------------------------------------------------------
__global__ void attn_scores(const float* __restrict__ blstm,
                            const float* __restrict__ aw,
                            const float* __restrict__ ab,
                            float* __restrict__ scores) {
  __shared__ float red[256];
  int s = blockIdx.x, tid = threadIdx.x;
  const float* row = blstm + (size_t)s * H2;
  float acc = 0.f;
  for (int c = tid; c < H2; c += 256) acc += row[c] * aw[c];
  red[tid] = acc;
  __syncthreads();
  for (int off = 128; off; off >>= 1) {
    if (tid < off) red[tid] += red[tid + off];
    __syncthreads();
  }
  if (tid == 0) scores[s] = red[0] + ab[0];
}

// ---------------------------------------------------------------------------
// Entity attention: softmax over 8 span scores -> weight sum -> scale row i.
// ---------------------------------------------------------------------------
__global__ void entity_attend(const int* __restrict__ spans,
                              const float* __restrict__ scores,
                              const float* __restrict__ blstm,
                              float* __restrict__ Hc) {
  __shared__ float wsum_s;
  __shared__ int anyv_s;
  int i = blockIdx.x, tid = threadIdx.x;
  if (tid == 0) {
    float g[8]; int msk[8]; int anyv = 0; float mx = -1e30f;
    for (int k = 0; k < 8; ++k) {
      int sp = spans[i * 8 + k];
      msk[k] = sp >= 0;
      anyv |= msk[k];
      int ci = sp < 0 ? 0 : (sp > SEQ - 1 ? SEQ - 1 : sp);
      g[k] = msk[k] ? scores[ci] : -1e30f;
      mx = fmaxf(mx, g[k]);
    }
    float den = 0.f;
    for (int k = 0; k < 8; ++k) den += __expf(g[k] - mx);
    float ws = 0.f;
    for (int k = 0; k < 8; ++k) if (msk[k]) ws += __expf(g[k] - mx) / den;
    wsum_s = ws; anyv_s = anyv;
  }
  __syncthreads();
  if (anyv_s) {
    float ws = wsum_s;
    for (int c = tid; c < H2; c += 256)
      Hc[(size_t)i * H2 + c] = ws * blstm[(size_t)i * H2 + c];
  }
}

// ---------------------------------------------------------------------------
// DAG-LSTM sequential scan: 1 block, 1024 threads, 1024 iterations.
// e_gates (elem_emb @ W_e + b_e) is precomputed by the WMMA GEMM.
// ---------------------------------------------------------------------------
__global__ void __launch_bounds__(1024)
dag_scan(const int* __restrict__ S, const int* __restrict__ T,
         const int* __restrict__ enames,
         const float* __restrict__ eg,     // [NREL, 4*H2]
         const float* __restrict__ Uiou,   // [2, H2, 3*H2]
         const float* __restrict__ Uf,     // [2, H2, H2]
         const float* __restrict__ w1, const float* __restrict__ b1,
         const float* __restrict__ w2, const float* __restrict__ b2,
         float* __restrict__ Hc, float* __restrict__ out) {
  __shared__ float preds[NTOT * 2];
  __shared__ float c[H2], hv0[H2], hv1[H2], hout[H2];
  __shared__ float lg[2];
  __shared__ int condS, validS, takeS;
  int tid = threadIdx.x;
  for (int idx = tid; idx < NTOT * 2; idx += 1024) {
    int r = idx >> 1;
    preds[idx] = (r < NENT) ? ((idx & 1) ? 0.999f : 0.001f) : 0.f;
  }
  c[tid] = 0.f;
  __syncthreads();

  for (int r = 0; r < NREL; ++r) {
    int a0 = S[2 * r], a1 = S[2 * r + 1], tgt = T[r], en = enames[r];
    hv0[tid] = Hc[(size_t)a0 * H2 + tid];
    hv1[tid] = Hc[(size_t)a1 * H2 + tid];
    if (tid == 0) {
      int cond = (preds[a0 * 2 + 1] > 0.5f) && (preds[a1 * 2 + 1] > 0.5f);
      int valid = (tgt >= NENT) && (en > -1);
      condS = cond; validS = valid; takeS = cond && valid;
      lg[0] = 0.f; lg[1] = 0.f;
    }
    __syncthreads();

    const float* er = eg + (size_t)r * WE_N;
    float gi = er[tid], go = er[H2 + tid], gu = er[2 * H2 + tid];
    for (int h = 0; h < H2; ++h) {
      float x0 = hv0[h], x1 = hv1[h];
      const float* u0 = Uiou + (size_t)h * 3 * H2;
      const float* u1 = Uiou + (size_t)(H2 + h) * 3 * H2;
      gi += x0 * u0[tid]          + x1 * u1[tid];
      go += x0 * u0[H2 + tid]     + x1 * u1[H2 + tid];
      gu += x0 * u0[2 * H2 + tid] + x1 * u1[2 * H2 + tid];
    }
    float f0 = er[3 * H2 + tid], f1 = f0;
    for (int h = 0; h < H2; ++h) {
      f0 += hv0[h] * Uf[(size_t)h * H2 + tid];
      f1 += hv1[h] * Uf[(size_t)(H2 + h) * H2 + tid];
    }
    f0 = 1.f / (1.f + __expf(-f0));
    f1 = 1.f / (1.f + __expf(-f1));
    float ck = c[tid];
    float cn = 1.f / (1.f + __expf(-gi)) * tanhf(gu) + f0 * ck + f1 * ck;
    cn = fminf(fmaxf(cn, -50.f), 50.f);
    float ho = 1.f / (1.f + __expf(-go)) * tanhf(cn);
    ho = fminf(fmaxf(ho, -50.f), 50.f);
    hout[tid] = ho;
    __syncthreads();

    float l0 = 0.f, l1 = 0.f;
#pragma unroll
    for (int q = 0; q < 4; ++q) {
      int k = tid * 4 + q;
      float m1 = b1[k];
      for (int h = 0; h < H2; ++h) m1 += hout[h] * w1[(size_t)h * WE_N + k];
      l0 += m1 * w2[k * 2];
      l1 += m1 * w2[k * 2 + 1];
    }
    atomicAdd(&lg[0], l0);
    atomicAdd(&lg[1], l1);
    __syncthreads();

    if (tid == 0) {
      float z0 = lg[0] + b2[0], z1 = lg[1] + b2[1];
      float mx = fmaxf(z0, z1);
      float e0 = __expf(z0 - mx), e1 = __expf(z1 - mx), den = e0 + e1;
      float p0 = condS ? e0 / den : 0.999f;
      float p1 = condS ? e1 / den : 0.001f;
      if (validS) { preds[tgt * 2] = p0; preds[tgt * 2 + 1] = p1; }
    }
    if (takeS) {
      Hc[(size_t)tgt * H2 + tid] = ho;
      c[tid] = cn;
    }
    __syncthreads();
  }
  for (int idx = tid; idx < NTOT * 2; idx += 1024) out[idx] = preds[idx];
}

// ---------------------------------------------------------------------------
// Host launcher
// ---------------------------------------------------------------------------
extern "C" void kernel_launch(void* const* d_in, const int* in_sizes, int n_in,
                              void* d_out, int out_size, void* d_ws, size_t ws_size,
                              hipStream_t stream) {
  (void)in_sizes; (void)n_in; (void)out_size; (void)ws_size;
  const int*   tokens   = (const int*)d_in[0];
  const int*   spans    = (const int*)d_in[1];
  const int*   enames   = (const int*)d_in[2];
  const float* Hin      = (const float*)d_in[3];
  const int*   Tarr     = (const int*)d_in[5];
  const int*   Sarr     = (const int*)d_in[6];
  const float* word_emb = (const float*)d_in[7];
  const float* elem_emb = (const float*)d_in[8];
  const float* attn_w   = (const float*)d_in[9];
  const float* attn_b   = (const float*)d_in[10];
  const float* w_ih_f   = (const float*)d_in[11];
  const float* w_hh_f   = (const float*)d_in[12];
  const float* b_f      = (const float*)d_in[13];
  const float* w_ih_b   = (const float*)d_in[14];
  const float* w_hh_b   = (const float*)d_in[15];
  const float* b_b      = (const float*)d_in[16];
  const float* W_e      = (const float*)d_in[17];
  const float* b_e      = (const float*)d_in[18];
  const float* U_iou    = (const float*)d_in[19];
  const float* U_f      = (const float*)d_in[20];
  const float* out_w1   = (const float*)d_in[21];
  const float* out_b1   = (const float*)d_in[22];
  const float* out_w2   = (const float*)d_in[23];
  const float* out_b2   = (const float*)d_in[24];
  float* out = (float*)d_out;

  char* base = (char*)d_ws;
  size_t off = 0;
  auto alloc = [&](size_t bytes) -> char* {
    char* p = base + off;
    off += (bytes + 255) & ~(size_t)255;
    return p;
  };
  float*    embF   = (float*)   alloc((size_t)SEQ * EMB * 4);
  _Float16* ApEmb  = (_Float16*)alloc((size_t)SEQ * EMB * 2);
  _Float16* BpWf   = (_Float16*)alloc((size_t)G4 * EMB * 2);
  _Float16* BpWb   = (_Float16*)alloc((size_t)G4 * EMB * 2);
  float*    XgF    = (float*)   alloc((size_t)SEQ * G4 * 4);
  float*    XgB    = (float*)   alloc((size_t)SEQ * G4 * 4);
  float*    blstm  = (float*)   alloc((size_t)SEQ * H2 * 4);
  float*    scores = (float*)   alloc((size_t)SEQ * 4);
  float*    EgF    = (float*)   alloc((size_t)NREL * REL * 4);
  _Float16* ApEg   = (_Float16*)alloc((size_t)NREL * REL * 2);
  _Float16* BpWe   = (_Float16*)alloc((size_t)REL * WE_N * 2);
  float*    egates = (float*)   alloc((size_t)NREL * WE_N * 4);
  float*    Hc     = (float*)   alloc((size_t)NTOT * H2 * 4);

  const int TB = 256;
  // 1) gathers
  gather_rows<<<(SEQ * EMB + TB - 1) / TB, TB, 0, stream>>>(
      tokens, word_emb, embF, SEQ, EMB, VOCAB);
  gather_rows<<<(NREL * REL + TB - 1) / TB, TB, 0, stream>>>(
      enames, elem_emb, EgF, NREL, REL, NELEM);
  // 2) fragment packing
  pack_a16<<<(SEQ * EMB + TB - 1) / TB, TB, 0, stream>>>(embF, ApEmb, SEQ, EMB);
  pack_a16<<<(NREL * REL + TB - 1) / TB, TB, 0, stream>>>(EgF, ApEg, NREL, REL);
  pack_b16<<<(G4 * EMB + TB - 1) / TB, TB, 0, stream>>>(w_ih_f, BpWf, EMB, G4, 1);
  pack_b16<<<(G4 * EMB + TB - 1) / TB, TB, 0, stream>>>(w_ih_b, BpWb, EMB, G4, 1);
  pack_b16<<<(REL * WE_N + TB - 1) / TB, TB, 0, stream>>>(W_e, BpWe, REL, WE_N, 0);
  // 3) register-blocked WMMA GEMMs (8 waves/block; wave-tile counts divisible
  //    by 8 -> EXEC all-ones at every WMMA)
  {
    int MT4 = (SEQ / 16) / 4, NT4 = (G4 / 16) / 4, KT = EMB / 32;  // 64, 32, 16
    int blocks = (MT4 * NT4) / 8;                                  // 256
    wmma_gemm4x4<<<blocks, TB, 0, stream>>>(ApEmb, BpWf, XgF, nullptr, NT4, KT, G4);
    wmma_gemm4x4<<<blocks, TB, 0, stream>>>(ApEmb, BpWb, XgB, nullptr, NT4, KT, G4);
  }
  {
    int MT4 = (NREL / 16) / 4, NT4 = (WE_N / 16) / 4, KT = REL / 32; // 16, 64, 8
    int blocks = (MT4 * NT4) / 8;                                    // 128
    wmma_gemm4x4<<<blocks, TB, 0, stream>>>(ApEg, BpWe, egates, b_e, NT4, KT, WE_N);
  }
  // 4) BiLSTM recurrences (independent directions)
  lstm_scan<<<1, 1024, 0, stream>>>(XgF, w_hh_f, b_f, blstm, 0);
  lstm_scan<<<1, 1024, 0, stream>>>(XgB, w_hh_b, b_b, blstm, 1);
  // 5) attention + entity rows; H -> Hc staged through the Tensor Data Mover
  attn_scores<<<SEQ, TB, 0, stream>>>(blstm, attn_w, attn_b, scores);
  tdm_copy_2d<<<NTOT / TDM_ROWS, 32, 0, stream>>>(Hin, Hc, NTOT, H2);
  entity_attend<<<NENT, TB, 0, stream>>>(spans, scores, blstm, Hc);
  // 6) DAG-LSTM sequential scan -> predictions
  dag_scan<<<1, 1024, 0, stream>>>(Sarr, Tarr, enames, egates, U_iou, U_f,
                                   out_w1, out_b1, out_w2, out_b2, Hc, out);
}
